// ConvAtt_9225589752077
// MI455X (gfx1250) — compile-verified, matrix operand order checked by reading, not run on hardware
//
#include <hip/hip_runtime.h>
#include <hip/hip_bf16.h>
#include <math.h>

typedef __attribute__((ext_vector_type(16))) _Float16 v16h;
typedef __attribute__((ext_vector_type(8)))  _Float16 v8h;
typedef __attribute__((ext_vector_type(8)))  float    v8f;

#define B_      16
#define CIN_    64
#define ATT_    16
#define H_      192
#define W_      192
#define HW_     (H_*W_)
#define LK_     13
#define NTAPS   169      // 13*13
#define NPAIRS  85       // ceil(169/2), tap 169 zero-padded
#define TILE_W  64
#define HALO    6
#define IN_COLS (TILE_W + 2*HALO)   // 76
#define IN_ROWS LK_                 // 13
#define IN_ROWS_P (IN_ROWS + 1)     // +1 zeroed row so padded tap 169 (kh=13) reads zeros

// ---------------- stage 1: global average pool over F_att ----------------
__global__ void pool_kernel(const float* __restrict__ x, float* __restrict__ pooled) {
    int bc = blockIdx.x;                     // b*16 + c , c < ATT
    const float* p = x + (size_t)((bc / 16) * CIN_ + (bc % 16)) * HW_;
    float s = 0.f;
    for (int i = threadIdx.x; i < HW_; i += blockDim.x) s += p[i];
    __shared__ float red[256];
    red[threadIdx.x] = s;
    __syncthreads();
    for (int off = 128; off > 0; off >>= 1) {
        if ((int)threadIdx.x < off) red[threadIdx.x] += red[threadIdx.x + off];
        __syncthreads();
    }
    if (threadIdx.x == 0) pooled[bc] = red[0] * (1.0f / (float)HW_);
}

// ---------------- stage 2: MLP -> dynamic 3x3 kernels (exact GELU) ----------------
__global__ void mlp_kernel(const float* __restrict__ pooled,
                           const float* __restrict__ w1, const float* __restrict__ b1,
                           const float* __restrict__ w2, const float* __restrict__ b2,
                           float* __restrict__ kdyn) {
    for (int idx = threadIdx.x; idx < B_ * 144; idx += blockDim.x) {
        int b = idx / 144, o = idx % 144;
        float acc = b2[o];
        for (int hh = 0; hh < 8; ++hh) {
            float s = b1[hh];
            for (int c = 0; c < 16; ++c) s += pooled[b * 16 + c] * w1[hh * 16 + c];
            float g = 0.5f * s * (1.0f + erff(s * 0.70710678118654752f));
            acc += g * w2[o * 8 + hh];
        }
        kdyn[idx] = acc;   // kdyn[b][oc][3][3] flat
    }
}

// ---------------- stage 3: pre-swizzle filters into WMMA A-fragment order ----------------
// A-frag 16-bit layout (16x32): lane l (g=l/16, oc=l%16), half j in 0..15:
//   K = 16*(j/8) + 8*g + (j%8)
__global__ void prep_filters(const float* __restrict__ lk, const float* __restrict__ fw,
                             _Float16* __restrict__ Aprep, _Float16* __restrict__ Afus) {
    int idx = blockIdx.x * blockDim.x + threadIdx.x;
    if (idx < NPAIRS * 512) {        // conv filter: 85 pairs x 32 lanes x 16 halfs
        int p = idx >> 9;
        int r = idx & 511;
        int l = r >> 4, j = r & 15;
        int g = l >> 4, oc = l & 15;
        int K = ((j >> 3) << 4) + (g << 3) + (j & 7);
        int tap = 2 * p + (K >> 4);
        int ic  = K & 15;
        float v = 0.f;
        if (tap < NTAPS)
            v = lk[((oc * 16 + ic) * LK_ + tap / LK_) * LK_ + tap % LK_];
        Aprep[idx] = (_Float16)v;
    }
    if (idx < 8 * 512) {             // fusion: 4 M-tiles x 2 K-chunks x 32 lanes x 16 halfs
        int f = idx >> 9;            // m*2+q
        int m = f >> 1, q = f & 1;
        int r = idx & 511;
        int l = r >> 4, j = r & 15;
        int o = m * 16 + (l & 15);
        int c = q * 32 + ((j >> 3) << 4) + ((l >> 4) << 3) + (j & 7);
        Afus[idx] = (_Float16)fw[o * 64 + c];
    }
}

// ---------------- stage 4: fused 13x13 conv + dynamic conv + channel fusion ----------------
__global__ __launch_bounds__(128)
void conv_fuse_kernel(const float* __restrict__ x,
                      const _Float16* __restrict__ Aprep,
                      const _Float16* __restrict__ Afus,
                      const float* __restrict__ kdyn,
                      const float* __restrict__ fusion_b,
                      float* __restrict__ out) {
    __shared__ _Float16 s_in[IN_ROWS_P * IN_COLS * 16];  // [row][col][ic], row 13 = zeros
    __shared__ _Float16 s_u[64 * 64];                    // [pixel][channel]

    const int tile = blockIdx.x;            // 0..2
    const int h    = blockIdx.y;            // 0..191
    const int b    = blockIdx.z;            // 0..15
    const int w0   = tile * TILE_W;
    const int tid  = threadIdx.x;
    const float* xb = x + (size_t)b * CIN_ * HW_;

    // ---- phase 1: stage zero-padded f16 halo tile of F_att in LDS ----
    for (int idx = tid; idx < 16 * IN_ROWS * IN_COLS; idx += 128) {
        int col = idx % IN_COLS;
        int t   = idx / IN_COLS;
        int row = t % IN_ROWS;
        int ic  = t / IN_ROWS;
        int gr = h + row - HALO;
        int gc = w0 + col - HALO;
        float v = 0.f;
        if (gr >= 0 && gr < H_ && gc >= 0 && gc < W_)
            v = xb[(size_t)ic * HW_ + gr * W_ + gc];
        s_in[(row * IN_COLS + col) * 16 + ic] = (_Float16)v;
    }
    // zero the extra row (row 13): target of the zero-padded tap 169
    for (int idx = tid; idx < IN_COLS * 16; idx += 128)
        s_in[IN_ROWS * IN_COLS * 16 + idx] = (_Float16)0.0f;
    __syncthreads();

    const int wave = tid >> 5, lane = tid & 31;
    const int g = lane >> 4, n = lane & 15;
    const int px_base = wave * 16;
    const int colb = px_base + n;
    const int goff = g << 3;

    // branch-free B-fragment gather (pair 84 -> tap 169 -> kh=13 -> zero row)
    auto loadB = [&](int pair) -> v16h {
        int t0 = 2 * pair, t1 = t0 + 1;
        int kh0 = t0 / 13, kw0 = t0 - kh0 * 13;
        int kh1 = t1 / 13, kw1 = t1 - kh1 * 13;
        v16h bf;
        ((v8h*)&bf)[0] = *(const v8h*)&s_in[((kh0 * IN_COLS + colb + kw0) << 4) + goff];
        ((v8h*)&bf)[1] = *(const v8h*)&s_in[((kh1 * IN_COLS + colb + kw1) << 4) + goff];
        return bf;
    };
    auto loadA = [&](int pair) -> v16h {
        return *(const v16h*)(Aprep + (((pair * 32) + lane) << 4));
    };

    // ---- phase 2: dense 13x13 conv, 85-WMMA K-loop, 2-deep software pipeline ----
    // With pairs p and p+1 resident and p+2's loads in flight, every WMMA waits
    // only on loads issued >= 2 iterations ago (partial s_wait, no full drain).
    v8f acc = {};
    v16h a0 = loadA(0), b0 = loadB(0);
    v16h a1 = loadA(1), b1v = loadB(1);
    #pragma unroll 2
    for (int p = 0; p < NPAIRS - 2; ++p) {
        v16h an = loadA(p + 2);
        v16h bn = loadB(p + 2);
        int pf = (p + 6 < NPAIRS) ? (p + 6) : (NPAIRS - 1);
        __builtin_prefetch(Aprep + (((pf * 32) + lane) << 4), 0, 1);
        acc = __builtin_amdgcn_wmma_f32_16x16x32_f16(false, a0, false, b0,
                                                     (short)0, acc, false, false);
        a0 = a1; b0 = b1v;
        a1 = an; b1v = bn;
    }
    acc = __builtin_amdgcn_wmma_f32_16x16x32_f16(false, a0, false, b0,
                                                 (short)0, acc, false, false);
    acc = __builtin_amdgcn_wmma_f32_16x16x32_f16(false, a1, false, b1v,
                                                 (short)0, acc, false, false);

    // ---- phase 3: add dynamic depthwise 3x3 conv ----
    // C/D layout: component v, lane group g -> oc = v + 8*g ; column n -> pixel
    const float* kd = kdyn + b * 144;
    #pragma unroll
    for (int v = 0; v < 8; ++v) {
        int oc = goff + v;
        const float* kk = kd + oc * 9;
        float s = 0.f;
        #pragma unroll
        for (int i = 0; i < 3; ++i)
            #pragma unroll
            for (int j = 0; j < 3; ++j) {
                int row = (HALO - 1) + i;
                int col = colb + (HALO - 1) + j;
                s += (float)s_in[((row * IN_COLS + col) << 4) + oc] * kk[i * 3 + j];
            }
        acc[v] += s;
    }

    // ---- phase 4: stage U = [conv result (ch 0..15) | F_idt (ch 16..63)] in LDS ----
    {
        v8h t;
        #pragma unroll
        for (int v = 0; v < 8; ++v) t[v] = (_Float16)acc[v];
        *(v8h*)&s_u[(colb << 6) + goff] = t;
    }
    for (int idx = tid; idx < 48 * 64; idx += 128) {
        int px = idx & 63;
        int c  = 16 + (idx >> 6);
        float v = xb[(size_t)c * HW_ + h * W_ + w0 + px];
        s_u[(px << 6) + c] = (_Float16)v;
    }
    __syncthreads();

    // ---- phase 5: 64x64 fusion GEMM (4 M-tiles x 2 K-chunks of WMMA) + bias ----
    #pragma unroll
    for (int m = 0; m < 4; ++m) {
        v8f facc = {};
        #pragma unroll
        for (int q = 0; q < 2; ++q) {
            v16h af = *(const v16h*)(Afus + ((((m * 2 + q) * 32) + lane) << 4));
            v16h bf;
            ((v8h*)&bf)[0] = *(const v8h*)&s_u[(colb << 6) + q * 32 + goff];
            ((v8h*)&bf)[1] = *(const v8h*)&s_u[(colb << 6) + q * 32 + 16 + goff];
            facc = __builtin_amdgcn_wmma_f32_16x16x32_f16(false, af, false, bf,
                                                          (short)0, facc, false, false);
        }
        #pragma unroll
        for (int v = 0; v < 8; ++v) {
            int o = m * 16 + goff + v;
            out[(((size_t)b * CIN_ + o) * H_ + h) * W_ + w0 + colb] = facc[v] + fusion_b[o];
        }
    }
}

// ---------------- launcher ----------------
extern "C" void kernel_launch(void* const* d_in, const int* in_sizes, int n_in,
                              void* d_out, int out_size, void* d_ws, size_t ws_size,
                              hipStream_t stream) {
    const float* x  = (const float*)d_in[0];
    const float* w1 = (const float*)d_in[1];
    const float* b1 = (const float*)d_in[2];
    const float* w2 = (const float*)d_in[3];
    const float* b2 = (const float*)d_in[4];
    const float* lk = (const float*)d_in[5];
    const float* fw = (const float*)d_in[6];
    const float* fb = (const float*)d_in[7];
    float* out = (float*)d_out;

    char* ws = (char*)d_ws;
    float*    pooled = (float*)(ws);              //   256 f  ->  1,024 B
    float*    kdyn   = (float*)(ws + 1024);       //  2304 f  ->  9,216 B
    _Float16* Aprep  = (_Float16*)(ws + 10240);   // 43,520 h -> 87,040 B
    _Float16* Afus   = (_Float16*)(ws + 97280);   //  4,096 h ->  8,192 B  (end 105,472)

    pool_kernel <<<dim3(B_ * ATT_), dim3(256), 0, stream>>>(x, pooled);
    mlp_kernel  <<<dim3(1),         dim3(256), 0, stream>>>(pooled, w1, b1, w2, b2, kdyn);
    {
        int tot = NPAIRS * 512;   // also covers the 8*512 fusion entries
        prep_filters<<<dim3((tot + 255) / 256), dim3(256), 0, stream>>>(lk, fw, Aprep, Afus);
    }
    conv_fuse_kernel<<<dim3(W_ / TILE_W, H_, B_), dim3(128), 0, stream>>>(
        x, Aprep, Afus, kdyn, fb, out);
}
